// MaterialEncoder_20796231647233
// MI455X (gfx1250) — matrix operand order repaired and checked by reading, not compiled, not run on hardware
//
#include <hip/hip_runtime.h>
#include <hip/hip_fp16.h>
#include <math.h>

typedef __attribute__((ext_vector_type(16))) _Float16 v16h;
typedef __attribute__((ext_vector_type(8)))  float    v8f;

#define WPB 8            // waves per block (wave32)
#define NT  256          // threads per block
#define FIN 83           // mat_feature_len
#define KP  96           // padded K (3 chunks of 32)
#define DD  32           // hidden/out dim

union HF { v16h v; _Float16 h[16]; unsigned u[8]; };

__device__ __forceinline__ void wave_ds_fence() {
    // DS ops are in-order within a wave; draining DScnt makes all lanes'
    // LDS stores visible to subsequent LDS loads of the same wave.
    asm volatile("s_wait_dscnt 0" ::: "memory");
}

// Exact GELU with branch-free erf (Abramowitz & Stegun 7.1.26, |err| < 1.5e-7).
// Uses raw v_rcp_f32 / v_exp_f32 (TRANS32 ops co-execute with VALU on CDNA5),
// no exec-mask branches -- ~15 straight-line ops vs ~40+ branchy ocml erff.
__device__ __forceinline__ float gelu_exact(float x) {
    const float u = x * 0.70710678118654752440f;      // x / sqrt(2)
    const float a = __builtin_fabsf(u);
    const float t = __builtin_amdgcn_rcpf(__builtin_fmaf(0.3275911f, a, 1.0f));
    float p = __builtin_fmaf(1.061405429f, t, -1.453152027f);
    p = __builtin_fmaf(p, t, 1.421413741f);
    p = __builtin_fmaf(p, t, -0.284496736f);
    p = __builtin_fmaf(p, t, 0.254829592f);
    p = p * t;
    const float e = __builtin_amdgcn_exp2f(-1.4426950408889634f * (u * u));
    float erfv = __builtin_fmaf(-p, e, 1.0f);         // erf(|u|)
    erfv = __builtin_copysignf(erfv, u);
    return 0.5f * x * (1.0f + erfv);
}

__global__ __launch_bounds__(NT) void matenc_wmma_kernel(
    const float* __restrict__ inp, const float* __restrict__ shiftp,
    const float* __restrict__ W0, const float* __restrict__ b0,
    const float* __restrict__ W1, const float* __restrict__ b1,
    const float* __restrict__ W2, const float* __restrict__ b2,
    float* __restrict__ out, float* __restrict__ outMask, int tiles)
{
    __shared__ _Float16 ldsX[WPB][16 * KP];
    __shared__ unsigned ldsM[WPB];

    const int lane = threadIdx.x & 31;
    const int wave = threadIdx.x >> 5;
    const int nloc = lane & 15;                // N column within a 16-wide tile
    const int hoff = (lane < 16) ? 0 : 8;      // A-frag K offset / D-frag row offset
    _Float16* lx = &ldsX[wave][0];

    const float shift = shiftp[0];

    // ---------------- Build B (weight) fragments once per wave ----------------
    // B layout used here (B is 32x16; contiguous K, same order as the A loads):
    //   lanes 0-15 hold K=0..15, lanes 16-31 hold K=16..31; VGPR j packs K=2j,2j+1.
    //   B[k][n] = W[n][k] (W stored [fan_out][fan_in] row-major).
    const int khB = (lane < 16) ? 0 : 16;
    HF B0f[3][2], B1f[2], B2f[2];
#pragma unroll
    for (int c = 0; c < 3; ++c)
#pragma unroll
        for (int t = 0; t < 2; ++t) {
            const int n = t * 16 + nloc;
#pragma unroll
            for (int j = 0; j < 8; ++j) {
                const int k0 = c * 32 + khB + 2 * j;
                B0f[c][t].h[2*j]   = (k0     < FIN) ? (_Float16)W0[n * FIN + k0    ] : (_Float16)0.0f;
                B0f[c][t].h[2*j+1] = (k0 + 1 < FIN) ? (_Float16)W0[n * FIN + k0 + 1] : (_Float16)0.0f;
            }
        }
#pragma unroll
    for (int t = 0; t < 2; ++t) {
        const int n = t * 16 + nloc;
#pragma unroll
        for (int j = 0; j < 8; ++j) {
            const int k0 = khB + 2 * j;
            B1f[t].h[2*j]   = (_Float16)W1[n * DD + k0];
            B1f[t].h[2*j+1] = (_Float16)W1[n * DD + k0 + 1];
            B2f[t].h[2*j]   = (_Float16)W2[n * DD + k0];
            B2f[t].h[2*j+1] = (_Float16)W2[n * DD + k0 + 1];
        }
    }
    const float bia0[2] = { b0[nloc], b0[16 + nloc] };
    const float bia1[2] = { b1[nloc], b1[16 + nloc] };
    const float bia2[2] = { b2[nloc], b2[16 + nloc] };

    const int waveGlobal = blockIdx.x * WPB + wave;
    const int waveStride = gridDim.x * WPB;

    for (int tile = waveGlobal; tile < tiles; tile += waveStride) {
        const int r0 = tile * 16;
        const float* src = inp + (size_t)r0 * FIN;

        if (lane == 0) ldsM[wave] = 0u;

        if (tile + waveStride < tiles)
            __builtin_prefetch(inp + (size_t)(tile + waveStride) * 16 * FIN, 0, 3);

        // ---- stage 16x83 input tile -> LDS f16 with shift substitution + mask ----
        unsigned lmask = 0u;
        for (int i = lane; i < 16 * FIN; i += 32) {
            const float v  = src[i];
            const int row  = i / FIN;
            const int col  = i - row * FIN;
            if (v != 0.0f) lmask |= (1u << row);
            lx[row * KP + col] = (_Float16)((v == 0.0f) ? shift : v);
        }
#pragma unroll 7
        for (int j = lane; j < 16 * (KP - FIN); j += 32) {        // zero K pad 83..95
            const int row = j / (KP - FIN);
            const int col = FIN + (j - row * (KP - FIN));
            lx[row * KP + col] = (_Float16)0.0f;
        }
        atomicOr(&ldsM[wave], lmask);
        wave_ds_fence();
        const unsigned maskbits = ldsM[wave];

        // ---- layer 0: [16x96]f16 @ B0 -> [16x32]f32, 3 K-chunks x 2 N-tiles ----
        v8f c0 = {}, c1 = {};
#pragma unroll
        for (int c = 0; c < 3; ++c) {
            HF a;
#pragma unroll
            for (int j = 0; j < 8; ++j) {
                const int k = c * 32 + 2 * hoff + 2 * j;
                a.u[j] = *(const unsigned*)&lx[nloc * KP + k];    // 4B-aligned pair
            }
            c0 = __builtin_amdgcn_wmma_f32_16x16x32_f16(false, a.v, false, B0f[c][0].v, (short)0, c0, false, false);
            c1 = __builtin_amdgcn_wmma_f32_16x16x32_f16(false, a.v, false, B0f[c][1].v, (short)0, c1, false, false);
        }
        // bias + exact GELU; reshape D-layout -> 16x32 f16 tile in LDS
#pragma unroll
        for (int r = 0; r < 8; ++r) {
            const int mr = r + hoff;
            lx[mr * DD + nloc]      = (_Float16)gelu_exact(c0[r] + bia0[0]);
            lx[mr * DD + 16 + nloc] = (_Float16)gelu_exact(c1[r] + bia0[1]);
        }
        wave_ds_fence();

        // ---- layer 1: [16x32] @ B1 ----
        v8f d0 = {}, d1 = {};
        {
            HF a;
#pragma unroll
            for (int j = 0; j < 8; ++j) {
                const int k = 2 * hoff + 2 * j;
                a.u[j] = *(const unsigned*)&lx[nloc * DD + k];
            }
            d0 = __builtin_amdgcn_wmma_f32_16x16x32_f16(false, a.v, false, B1f[0].v, (short)0, d0, false, false);
            d1 = __builtin_amdgcn_wmma_f32_16x16x32_f16(false, a.v, false, B1f[1].v, (short)0, d1, false, false);
        }
#pragma unroll
        for (int r = 0; r < 8; ++r) {
            const int mr = r + hoff;
            lx[mr * DD + nloc]      = (_Float16)gelu_exact(d0[r] + bia1[0]);
            lx[mr * DD + 16 + nloc] = (_Float16)gelu_exact(d1[r] + bia1[1]);
        }
        wave_ds_fence();

        // ---- layer 2: [16x32] @ B2, keep result in registers ----
        v8f e0 = {}, e1 = {};
        {
            HF a;
#pragma unroll
            for (int j = 0; j < 8; ++j) {
                const int k = 2 * hoff + 2 * j;
                a.u[j] = *(const unsigned*)&lx[nloc * DD + k];
            }
            e0 = __builtin_amdgcn_wmma_f32_16x16x32_f16(false, a.v, false, B2f[0].v, (short)0, e0, false, false);
            e1 = __builtin_amdgcn_wmma_f32_16x16x32_f16(false, a.v, false, B2f[1].v, (short)0, e1, false, false);
        }
        float f0[8], f1[8], ss[8];
#pragma unroll
        for (int r = 0; r < 8; ++r) {
            f0[r] = gelu_exact(e0[r] + bia2[0]);
            f1[r] = gelu_exact(e1[r] + bia2[1]);
            ss[r] = f0[r] * f0[r] + f1[r] * f1[r];
        }
        // Row m lives in one 16-lane half (VGPR r, rows r / r+8): xor-reduce
        // within the half gives the full 32-element sum-of-squares per row.
#pragma unroll
        for (int xm = 1; xm <= 8; xm <<= 1)
#pragma unroll
            for (int r = 0; r < 8; ++r) ss[r] += __shfl_xor(ss[r], xm, 32);

        // ---- L2 normalize, apply zero-row mask, store ----
#pragma unroll
        for (int r = 0; r < 8; ++r) {
            const int mr = r + hoff;
            const float rn = (ss[r] > 0.0f) ? __builtin_amdgcn_rsqf(ss[r]) : 0.0f;
            const float keep = ((maskbits >> mr) & 1u) ? rn : 0.0f;
            const size_t base = (size_t)(r0 + mr) * DD;
            out[base + nloc]      = f0[r] * keep;
            out[base + 16 + nloc] = f1[r] * keep;
        }
        if (outMask != nullptr && nloc == 0) {
#pragma unroll
            for (int r = 0; r < 8; ++r) {
                const int mr = r + hoff;
                outMask[r0 + mr] = ((maskbits >> mr) & 1u) ? 1.0f : 0.0f;
            }
        }
    }
}

extern "C" void kernel_launch(void* const* d_in, const int* in_sizes, int n_in,
                              void* d_out, int out_size, void* d_ws, size_t ws_size,
                              hipStream_t stream) {
    const float* inp   = (const float*)d_in[0];
    const float* shift = (const float*)d_in[1];
    const float* W0 = (const float*)d_in[2];
    const float* b0 = (const float*)d_in[3];
    const float* W1 = (const float*)d_in[4];
    const float* b1 = (const float*)d_in[5];
    const float* W2 = (const float*)d_in[6];
    const float* b2 = (const float*)d_in[7];
    float* out = (float*)d_out;

    const int n_rows = in_sizes[0] / FIN;
    const int tiles  = n_rows / 16;

    float* outMask = nullptr;
    if ((long long)out_size >= (long long)n_rows * (DD + 1))
        outMask = out + (size_t)n_rows * DD;   // tuple: [x | mask] concatenated flat

    int blocks = (tiles + WPB - 1) / WPB;
    if (blocks > 4096) blocks = 4096;          // grid-stride: ~2 tiles per wave
    if (blocks < 1) blocks = 1;

    matenc_wmma_kernel<<<blocks, NT, 0, stream>>>(inp, shift, W0, b0, W1, b1, W2, b2,
                                                  out, outMask, tiles);
}
